// SwinTransformerBlock_28226525070367
// MI455X (gfx1250) — compile-verified, matrix operand order checked by reading, not run on hardware
//
#include <hip/hip_runtime.h>
#include <hip/hip_bf16.h>
#include <math.h>

// ---------------------------------------------------------------------------
// Swin Transformer block for MI455X (gfx1250, wave32, WMMA 16x16x32 f16)
// B=16, H=W=64, C=384, ws=8 (no padding), heads=12, hd=32, N=64 tokens/window
// ---------------------------------------------------------------------------

typedef _Float16 v16h __attribute__((ext_vector_type(16)));
typedef _Float16 v8h  __attribute__((ext_vector_type(8)));
typedef float    v8f  __attribute__((ext_vector_type(8)));

#define LN_EPS 1e-5f

__device__ __forceinline__ v16h make_frag(v8h lo, v8h hi) {
  return __builtin_shufflevector(lo, hi, 0,1,2,3,4,5,6,7,8,9,10,11,12,13,14,15);
}

// Lane fragment load: 16 halves as two contiguous 16B spans (k .. k+7, k+16 .. k+23)
__device__ __forceinline__ v16h load_frag(const _Float16* p) {
  v8h lo = *(const v8h*)p;
  v8h hi = *(const v8h*)(p + 16);
  return make_frag(lo, hi);
}

// Packed-B fragment: 16 contiguous halves per lane
__device__ __forceinline__ v16h load_frag_packed(const _Float16* p) {
  v8h lo = *(const v8h*)p;
  v8h hi = *(const v8h*)(p + 8);
  return make_frag(lo, hi);
}

__device__ __forceinline__ float wave_sum(float v) {
#pragma unroll
  for (int o = 16; o > 0; o >>= 1) v += __shfl_xor(v, o, 32);
  return v;
}

__device__ __forceinline__ v8f wmma_f16(v16h a, v16h b, v8f c) {
  return __builtin_amdgcn_wmma_f32_16x16x32_f16(false, a, false, b, (short)0, c,
                                                false, false);
}

// ---------------------------------------------------------------------------
// Weight repack: f32 [K,N] row-major -> f16 fragment-major [nt][kt][lane][16]
// Lane L holds column n = nt*16 + (L&15); halves 0..7 : k = kt*32 + (L>>4)*8 + e
//                                         halves 8..15: k = kt*32 + 16 + (L>>4)*8 + e
// ---------------------------------------------------------------------------
__global__ void pack_b_kernel(const float* __restrict__ B, _Float16* __restrict__ Bp,
                              int Kdim, int Ndim) {
  int tid  = blockIdx.x * blockDim.x + threadIdx.x;
  int lane = tid & 31;
  int t    = tid >> 5;
  int Ktiles = Kdim >> 5;
  int kt = t % Ktiles;
  int nt = t / Ktiles;
  if (nt >= (Ndim >> 4)) return;
  int n  = nt * 16 + (lane & 15);
  int kb = kt * 32 + (lane >> 4) * 8;
  v16h f;
#pragma unroll
  for (int e = 0; e < 8; ++e) f[e]     = (_Float16)B[(size_t)(kb + e) * Ndim + n];
#pragma unroll
  for (int e = 0; e < 8; ++e) f[e + 8] = (_Float16)B[(size_t)(kb + 16 + e) * Ndim + n];
  *(v16h*)(Bp + ((size_t)(nt * Ktiles + kt) * 32 + lane) * 16) = f;
}

// ---------------------------------------------------------------------------
// LayerNorm over C=384; one wave per row. PARTITION=1: window-partition output.
// ---------------------------------------------------------------------------
template <int PARTITION>
__global__ __launch_bounds__(256) void ln_kernel(const float* __restrict__ x,
                                                 const float* __restrict__ w,
                                                 const float* __restrict__ b,
                                                 _Float16* __restrict__ out) {
  int wave = threadIdx.x >> 5, lane = threadIdx.x & 31;
  int r = blockIdx.x * 8 + wave;  // 0..65535
  size_t src;
  if constexpr (PARTITION) {
    int wnd = r >> 6, t = r & 63;
    int bb = wnd >> 6, wi = wnd & 63;
    int ih = (wi >> 3) * 8 + (t >> 3);
    int iw = (wi & 7) * 8 + (t & 7);
    src = (((size_t)bb * 64 + ih) * 64 + iw) * 384;
  } else {
    src = (size_t)r * 384;
  }
  float xv[12];
  float s = 0.f;
#pragma unroll
  for (int i = 0; i < 12; ++i) { xv[i] = x[src + lane + i * 32]; s += xv[i]; }
  float mean = wave_sum(s) * (1.f / 384.f);
  float vs = 0.f;
#pragma unroll
  for (int i = 0; i < 12; ++i) { float d = xv[i] - mean; vs += d * d; }
  float rstd = rsqrtf(wave_sum(vs) * (1.f / 384.f) + LN_EPS);
  size_t dst = (size_t)r * 384;
#pragma unroll
  for (int i = 0; i < 12; ++i) {
    int c = lane + i * 32;
    out[dst + c] = (_Float16)((xv[i] - mean) * rstd * w[c] + b[c]);
  }
}

// ---------------------------------------------------------------------------
// Generic WMMA GEMM: C[M,N] = A[M,K](f16 rowmajor) * Bpacked + bias, epilogue.
// Block = 256 threads = 8 waves (4 M x 2 N), wave tile 32x32, block tile 128x64.
// MODE 0: QKV scatter (q scaled, v transposed)   MODE 1: proj + residual unpartition
// MODE 2: mlp1 + exact GELU                      MODE 3: mlp2 + residual -> out
// ---------------------------------------------------------------------------
template <int MODE>
__global__ __launch_bounds__(256) void gemm_kernel(
    const _Float16* __restrict__ A, const _Float16* __restrict__ Bp,
    const float* __restrict__ bias, int Ksz,
    _Float16* __restrict__ oQ, _Float16* __restrict__ oK, _Float16* __restrict__ oV,
    const float* __restrict__ res, float* __restrict__ o32) {
  int wave = threadIdx.x >> 5, lane = threadIdx.x & 31;
  int waveM = wave & 3, waveN = wave >> 2;
  int mBase = blockIdx.x * 128 + waveM * 32;
  int nBase = blockIdx.y * 64 + waveN * 32;
  int Ktiles = Ksz >> 5;
  int half_ = lane >> 4;
  int rowLo = mBase + (lane & 15);

  v8f acc[2][2] = {};
  const _Float16* bp0 = Bp + ((size_t)((nBase >> 4)) * Ktiles * 32 + lane) * 16;
  const _Float16* bp1 = Bp + ((size_t)((nBase >> 4) + 1) * Ktiles * 32 + lane) * 16;
  const _Float16* pa = A + (size_t)rowLo * Ksz + half_ * 8;

  for (int kt = 0; kt < Ktiles; ++kt) {
    __builtin_prefetch((const void*)(pa + 32), 0, 0);
    v16h a0 = load_frag(pa);
    v16h a1 = load_frag(pa + (size_t)16 * Ksz);
    v16h b0 = load_frag_packed(bp0 + (size_t)kt * 32 * 16);
    v16h b1 = load_frag_packed(bp1 + (size_t)kt * 32 * 16);
    acc[0][0] = wmma_f16(a0, b0, acc[0][0]);
    acc[0][1] = wmma_f16(a0, b1, acc[0][1]);
    acc[1][0] = wmma_f16(a1, b0, acc[1][0]);
    acc[1][1] = wmma_f16(a1, b1, acc[1][1]);
    pa += 32;
  }

#pragma unroll
  for (int i = 0; i < 2; ++i)
#pragma unroll
    for (int j = 0; j < 2; ++j)
#pragma unroll
      for (int r = 0; r < 8; ++r) {
        int gm = mBase + i * 16 + half_ * 8 + r;
        int gn = nBase + j * 16 + (lane & 15);
        float v = acc[i][j][r] + bias[gn];
        if constexpr (MODE == 0) {
          int which = gn / 384, rem = gn % 384;
          int h = rem >> 5, d = rem & 31;
          int w = gm >> 6, t = gm & 63;
          size_t wh = (size_t)w * 12 + h;
          if (which == 0)
            oQ[(wh * 64 + t) * 32 + d] = (_Float16)(v * 0.17677669529663687f);
          else if (which == 1)
            oK[(wh * 64 + t) * 32 + d] = (_Float16)v;
          else
            oV[(wh * 32 + d) * 64 + t] = (_Float16)v;  // transposed [d][m]
        } else if constexpr (MODE == 1) {
          int w = gm >> 6, t = gm & 63;
          int bb = w >> 6, wi = w & 63;
          int ih = (wi >> 3) * 8 + (t >> 3);
          int iw = (wi & 7) * 8 + (t & 7);
          size_t idx = (((size_t)bb * 64 + ih) * 64 + iw) * 384 + gn;
          o32[idx] = res[idx] + v;
        } else if constexpr (MODE == 2) {
          float g = 0.5f * v * (1.f + erff(v * 0.7071067811865475f));
          oQ[(size_t)gm * 1536 + gn] = (_Float16)g;
        } else {
          size_t idx = (size_t)gm * 384 + gn;
          o32[idx] = res[idx] + v;
        }
      }
}

// ---------------------------------------------------------------------------
// Windowed attention: one wave per (window, head). 2 waves / block (48KB LDS).
// S = Q*K^T + rel-pos bias -> softmax -> P*V. Q pre-scaled by hd^-0.5.
// ---------------------------------------------------------------------------
__global__ __launch_bounds__(64) void attn_kernel(const _Float16* __restrict__ Q,
                                                  const _Float16* __restrict__ K,
                                                  const _Float16* __restrict__ Vt,
                                                  const float* __restrict__ rpb,
                                                  _Float16* __restrict__ Ab) {
  __shared__ __align__(16) float Sbuf[2][64 * 64];
  __shared__ __align__(16) _Float16 Pbuf[2][64 * 64];
  int slot = threadIdx.x >> 5, lane = threadIdx.x & 31;
  int wh = blockIdx.x * 2 + slot;
  int w = wh / 12, h = wh % 12;
  int half_ = lane >> 4, lo = lane & 15;
  const _Float16* Qw = Q + (size_t)wh * 2048;
  const _Float16* Kw = K + (size_t)wh * 2048;
  const _Float16* Vw = Vt + (size_t)wh * 2048;
  float* S = Sbuf[slot];
  _Float16* P = Pbuf[slot];

  // S = Q (64x32) * K^T (32x64): B-fragment of K^T == A-style load of K rows.
  v16h bk[4];
#pragma unroll
  for (int nt = 0; nt < 4; ++nt)
    bk[nt] = load_frag(Kw + (size_t)(nt * 16 + lo) * 32 + half_ * 8);
#pragma unroll
  for (int mt = 0; mt < 4; ++mt) {
    v16h aq = load_frag(Qw + (size_t)(mt * 16 + lo) * 32 + half_ * 8);
#pragma unroll
    for (int nt = 0; nt < 4; ++nt) {
      v8f s = {};
      s = wmma_f16(aq, bk[nt], s);
#pragma unroll
      for (int r = 0; r < 8; ++r) {
        int m = mt * 16 + half_ * 8 + r;  // query token
        int n = nt * 16 + lo;             // key token
        int idx = ((m >> 3) - (n >> 3) + 7) * 15 + ((m & 7) - (n & 7) + 7);
        S[m * 64 + n] = s[r] + rpb[idx * 12 + h];
      }
    }
  }
  __syncthreads();

  // softmax: each lane owns rows 2*lane, 2*lane+1
#pragma unroll
  for (int rr = 0; rr < 2; ++rr) {
    int row = lane * 2 + rr;
    float* sr = S + row * 64;
    float mx = -3.4e38f;
    for (int n = 0; n < 64; ++n) mx = fmaxf(mx, sr[n]);
    float sum = 0.f;
    for (int n = 0; n < 64; ++n) {
      float e = expf(sr[n] - mx);
      sr[n] = e;
      sum += e;
    }
    float inv = 1.f / sum;
    _Float16* pr = P + row * 64;
    for (int n = 0; n < 64; ++n) pr[n] = (_Float16)(sr[n] * inv);
  }
  __syncthreads();

  // Out = P (64x64) * V (64x32); V stored transposed so B frags are contiguous.
  v8f o[4][2] = {};
#pragma unroll
  for (int kt = 0; kt < 2; ++kt) {
    v16h bv[2];
#pragma unroll
    for (int j = 0; j < 2; ++j)
      bv[j] = load_frag(Vw + (size_t)(j * 16 + lo) * 64 + kt * 32 + half_ * 8);
#pragma unroll
    for (int mt = 0; mt < 4; ++mt) {
      v16h ap = load_frag(P + (size_t)(mt * 16 + lo) * 64 + kt * 32 + half_ * 8);
      o[mt][0] = wmma_f16(ap, bv[0], o[mt][0]);
      o[mt][1] = wmma_f16(ap, bv[1], o[mt][1]);
    }
  }
#pragma unroll
  for (int mt = 0; mt < 4; ++mt)
#pragma unroll
    for (int j = 0; j < 2; ++j)
#pragma unroll
      for (int r = 0; r < 8; ++r) {
        int m = mt * 16 + half_ * 8 + r;
        int d = j * 16 + lo;
        Ab[((size_t)w * 64 + m) * 384 + h * 32 + d] = (_Float16)o[mt][j][r];
      }
}

// ---------------------------------------------------------------------------
extern "C" void kernel_launch(void* const* d_in, const int* in_sizes, int n_in,
                              void* d_out, int out_size, void* d_ws, size_t ws_size,
                              hipStream_t stream) {
  const float* x      = (const float*)d_in[0];
  const float* ln1_w  = (const float*)d_in[1];
  const float* ln1_b  = (const float*)d_in[2];
  const float* qkv_w  = (const float*)d_in[3];
  const float* qkv_b  = (const float*)d_in[4];
  const float* proj_w = (const float*)d_in[5];
  const float* proj_b = (const float*)d_in[6];
  const float* rpb    = (const float*)d_in[7];
  const float* ln2_w  = (const float*)d_in[8];
  const float* ln2_b  = (const float*)d_in[9];
  const float* mlp_w1 = (const float*)d_in[10];
  const float* mlp_b1 = (const float*)d_in[11];
  const float* mlp_w2 = (const float*)d_in[12];
  const float* mlp_b2 = (const float*)d_in[13];
  (void)in_sizes; (void)n_in; (void)out_size; (void)ws_size;

  char* ws = (char*)d_ws;
  // workspace layout (bytes)
  float*    X1 = (float*)(ws + 0ull);                 // 100663296 B  (x + attn residual)
  _Float16* Xw = (_Float16*)(ws + 100663296ull);      //  50331648 B  (LN1 windows; reused as Ab, H2)
  _Float16* Qb = (_Float16*)(ws + 150994944ull);      //  50331648 B
  _Float16* Kb = (_Float16*)(ws + 201326592ull);      //  50331648 B
  _Float16* Vt = (_Float16*)(ws + 251658240ull);      //  50331648 B  (+50331648 B spare region)
  _Float16* M1 = Qb;                                  // 201326592 B  (aliases Q..spare, dead by MLP)
  _Float16* Ab = Xw;                                  // attention output (window layout)
  _Float16* H2 = Xw;                                  // LN2 output (token layout)
  _Float16* Wq = (_Float16*)(ws + 352321536ull);      //   884736 B
  _Float16* Wp = (_Float16*)(ws + 353206272ull);      //   294912 B
  _Float16* W1 = (_Float16*)(ws + 353501184ull);      //  1179648 B
  _Float16* W2 = (_Float16*)(ws + 354680832ull);      //  1179648 B  (end: 355860480)

  // 1) pack weights to f16 fragment-major (L2-resident, ~3.5MB total)
  pack_b_kernel<<<108, 256, 0, stream>>>(qkv_w, Wq, 384, 1152);
  pack_b_kernel<<<36, 256, 0, stream>>>(proj_w, Wp, 384, 384);
  pack_b_kernel<<<144, 256, 0, stream>>>(mlp_w1, W1, 384, 1536);
  pack_b_kernel<<<144, 256, 0, stream>>>(mlp_w2, W2, 1536, 384);

  // 2) LN1 + window partition -> f16
  ln_kernel<1><<<8192, 256, 0, stream>>>(x, ln1_w, ln1_b, Xw);

  // 3) QKV GEMM (M=65536, K=384, N=1152), scatter to Q/K/V^T
  gemm_kernel<0><<<dim3(512, 18), 256, 0, stream>>>(Xw, Wq, qkv_b, 384, Qb, Kb, Vt,
                                                    nullptr, nullptr);

  // 4) windowed attention (12288 window-heads, 2 per block)
  attn_kernel<<<6144, 64, 0, stream>>>(Qb, Kb, Vt, rpb, Ab);

  // 5) proj GEMM + un-partition + residual -> X1 (f32)
  gemm_kernel<1><<<dim3(512, 6), 256, 0, stream>>>(Ab, Wp, proj_b, 384, nullptr,
                                                   nullptr, nullptr, x, X1);

  // 6) LN2 -> f16 (token-major)
  ln_kernel<0><<<8192, 256, 0, stream>>>(X1, ln2_w, ln2_b, H2);

  // 7) MLP1 GEMM + GELU -> M1 (M=65536, K=384, N=1536)
  gemm_kernel<2><<<dim3(512, 24), 256, 0, stream>>>(H2, W1, mlp_b1, 384, M1, nullptr,
                                                    nullptr, nullptr, nullptr);

  // 8) MLP2 GEMM + residual -> d_out (M=65536, K=1536, N=384)
  gemm_kernel<3><<<dim3(512, 6), 256, 0, stream>>>(M1, W2, mlp_b2, 1536, nullptr,
                                                   nullptr, nullptr, X1, (float*)d_out);
}
